// genconv_mg_9732395893310
// MI455X (gfx1250) — matrix operand order repaired and checked by reading, not compiled, not run on hardware
//
#include <hip/hip_runtime.h>
#include <hip/hip_bf16.h>
#include <math.h>

typedef _Float16 half_t;
typedef __attribute__((ext_vector_type(16))) _Float16 v16h;
typedef __attribute__((ext_vector_type(8)))  _Float16 v8h;
typedef __attribute__((ext_vector_type(8)))  float    v8f;

namespace {
constexpr int   N_NODES = 20000;
constexpr int   T_DIM   = 2;
constexpr int   H_DIM   = 128;
constexpr int   L_DIM   = 4;
constexpr int   E_DIM   = 320000;
constexpr int   F_DIM   = 514;
constexpr int   CN_DIM  = 3;
constexpr int   CG_DIM  = 2;
constexpr float EPS_MSG = 1e-7f;

constexpr int BM  = 32;   // block tile rows
constexpr int BN  = 64;   // block tile cols (4 waves x 16)
constexpr int BK  = 32;   // k-step = WMMA K
constexpr int LDA = 40;   // halves per A row in LDS (80B: 16B aligned, bank-spread)
constexpr int LDB = 40;   // halves per B column in LDS
} // namespace

// permute K index so that each lane's WMMA A-fragment is contiguous in LDS:
// positions {0..7,8..15,16..23,24..31} hold K {0..7,16..23,8..15,24..31}
// (swap bits 3 and 4 of k)
__device__ __forceinline__ int permK(int k) {
  return (k & ~24) | ((k & 8) << 1) | ((k & 16) >> 1);
}

// ---------------------------------------------------------------------------
// block-wide sum reduction (blockDim multiple of 32, <= 1024)
// ---------------------------------------------------------------------------
__device__ __forceinline__ float block_reduce_sum(float v, float* sh) {
  int lane = threadIdx.x & 31;
  int w    = threadIdx.x >> 5;
  #pragma unroll
  for (int o = 16; o > 0; o >>= 1) v += __shfl_down(v, o, 32);
  if (lane == 0) sh[w] = v;
  __syncthreads();
  int nw = (blockDim.x + 31) >> 5;
  if (w == 0) {
    float s = (lane < nw) ? sh[lane] : 0.0f;
    #pragma unroll
    for (int o = 16; o > 0; o >>= 1) s += __shfl_down(s, o, 32);
    if (lane == 0) sh[0] = s;
  }
  __syncthreads();
  float s = sh[0];
  __syncthreads();
  return s;
}

// ---------------------------------------------------------------------------
// WMMA GEMM: C[M,Nc] = A[M,K] @ B[K,Nc] (+bias) (+optional relu / accumulate)
// fp32 in/out, fp16 WMMA with f32 accumulation. 8 waves -> 32x64 block tile.
// LDS tiles stored in fragment-ready order so each lane reads its 16-half
// fragment with two ds_load_b128.
// ---------------------------------------------------------------------------
__global__ __launch_bounds__(256) void gemm_wmma_kernel(
    const float* __restrict__ A, const float* __restrict__ B,
    const float* __restrict__ bias, float* __restrict__ C,
    int M, int Nc, int K, int relu, int accum)
{
  __shared__ half_t lA[BM * LDA];   // [row][permuted k]
  __shared__ half_t lB[BN * LDB];   // [col][k]  (transposed: K contiguous)

  const int tid  = threadIdx.x;
  const int lane = tid & 31;
  const int wave = tid >> 5;
  const int wm   = wave >> 2;     // 0..1 : M sub-tile
  const int wn   = wave & 3;      // 0..3 : N sub-tile
  const int rowBase = blockIdx.y * BM;
  const int colBase = blockIdx.x * BN;

  // staging indices (full-tile fast path)
  const int ar  = tid >> 3;          // 0..31 : A row
  const int akq = (tid & 7) * 4;     // 0,4,..,28 : A k-quad (stays in 8-group)
  const int apk = permK(akq);        // contiguous 4-half store position
  const int bk  = tid >> 5;          // 0..7 : B k rows handled 4 at a time below
  const int bn0 = (lane) * 2;        // 0..62 : B col pair

  v8f acc = {};
  const int ksteps = (K + BK - 1) / BK;

  for (int kt = 0; kt < ksteps; ++kt) {
    const int k0g = kt * BK;
    const bool full = (k0g + BK <= K) & (rowBase + BM <= M);  // uniform
    if (full) {
      // A: one thread = 4 consecutive k of one row (2x b64 loads: rows are
      // only 8B-aligned when K==514)
      {
        const float* ap = A + (size_t)(rowBase + ar) * K + (k0g + akq);
        float2 p0 = *reinterpret_cast<const float2*>(ap);
        float2 p1 = *reinterpret_cast<const float2*>(ap + 2);
        half_t* dst = &lA[ar * LDA + apk];
        dst[0] = (half_t)p0.x; dst[1] = (half_t)p0.y;
        dst[2] = (half_t)p1.x; dst[3] = (half_t)p1.y;
      }
      // B: one thread = 2 consecutive n of 4 k rows k+8*kk (4x b64), stored
      // transposed into LDS (K contiguous per column)
      {
        int k = tid >> 5;              // 0..7
        #pragma unroll
        for (int kk = 0; kk < 4; ++kk) {
          int gk = k0g + k + kk * 8;
          const float* bp = B + (size_t)gk * Nc + colBase + lane * 2;
          float2 q = *reinterpret_cast<const float2*>(bp);
          lB[(lane * 2 + 0) * LDB + k + kk * 8] = (half_t)q.x;
          lB[(lane * 2 + 1) * LDB + k + kk * 8] = (half_t)q.y;
        }
      }
    } else {
      // ragged tail (only the last k-step of K=514): guarded scalar staging
      for (int e = tid; e < BM * BK; e += 256) {
        int r = e >> 5, k = e & 31;
        int gr = rowBase + r, gk = k0g + k;
        float v = (gr < M && gk < K) ? A[(size_t)gr * K + gk] : 0.0f;
        lA[r * LDA + permK(k)] = (half_t)v;
      }
      for (int e = tid; e < BK * BN; e += 256) {
        int k = e >> 6, c = e & 63;
        int gk = k0g + k, gc = colBase + c;
        float v = (gk < K && gc < Nc) ? B[(size_t)gk * Nc + gc] : 0.0f;
        lB[c * LDB + k] = (half_t)v;
      }
    }
    __syncthreads();

    // A fragment: contiguous 32B per lane -> 2x ds_load_b128
    const int r    = lane & 15;
    const int koff = (lane >> 4) * 16;
    const v8h* pa = reinterpret_cast<const v8h*>(&lA[(wm * 16 + r) * LDA + koff]);
    v8h a0 = pa[0], a1 = pa[1];
    v16h af = __builtin_shufflevector(a0, a1, 0, 1, 2, 3, 4, 5, 6, 7,
                                      8, 9, 10, 11, 12, 13, 14, 15);
    // B fragment: lanes 0-15 N=lane K=0..7,16..23; lanes 16-31 N=lane-16 K=8..15,24..31
    const v8h* pb = reinterpret_cast<const v8h*>(&lB[(wn * 16 + r) * LDB + (lane >> 4) * 8]);
    v8h b0 = pb[0], b1 = pb[2];
    v16h bf = __builtin_shufflevector(b0, b1, 0, 1, 2, 3, 4, 5, 6, 7,
                                      8, 9, 10, 11, 12, 13, 14, 15);

    acc = __builtin_amdgcn_wmma_f32_16x16x32_f16(
        false, af, false, bf, (short)0, acc, false, false);
    __syncthreads();
  }

  // epilogue: C/D layout: lane<16 -> N=lane, M = i (+8 for lanes>=16)
  const int nn    = lane & 15;
  const int mHalf = (lane >> 4) * 8;
  const int col   = colBase + wn * 16 + nn;
  #pragma unroll
  for (int i = 0; i < 8; ++i) {
    int row = rowBase + wm * 16 + mHalf + i;
    if (row < M && col < Nc) {
      float v = acc[i];
      if (bias)  v += bias[col];
      if (relu)  v = fmaxf(v, 0.0f);
      size_t idx = (size_t)row * Nc + col;
      if (accum) v += C[idx];
      C[idx] = v;
    }
  }
}

// ---------------------------------------------------------------------------
// segment-softmax edge passes: grid-stride persistent waves, one wave32 per
// edge (float4 per lane -> H=128), prefetch of next gather row.
// pass 0: atomic max of logits per dst
// pass 1: atomic add of exp(logit - m[dst]) into denom
// pass 2: atomic add of msg * exp(...) / (denom+1e-16) into agg
// ---------------------------------------------------------------------------
__device__ __forceinline__ void atomicMaxFloat(float* addr, float val) {
  unsigned int* ua  = reinterpret_cast<unsigned int*>(addr);
  unsigned int  cur = __float_as_uint(*addr);
  while (__uint_as_float(cur) < val) {
    unsigned int prev = atomicCAS(ua, cur, __float_as_uint(val));
    if (prev == cur) break;
    cur = prev;
  }
}

__global__ __launch_bounds__(256) void edge_pass_kernel(
    int pass,
    const float* __restrict__ x, const int* __restrict__ src,
    const int* __restrict__ dst, const float* __restrict__ ea,
    const float* __restrict__ Wee, const float* __restrict__ bee,
    const float* __restrict__ tptr, int layer,
    float* __restrict__ m, float* __restrict__ den, float* __restrict__ agg)
{
  const int lane = threadIdx.x & 31;
  const int h    = lane * 4;
  const int wpg  = gridDim.x * (blockDim.x >> 5);

  const float tv  = tptr[layer];
  const float4 w0 = *reinterpret_cast<const float4*>(Wee + h);
  const float4 w1 = *reinterpret_cast<const float4*>(Wee + H_DIM + h);
  const float4 bb = *reinterpret_cast<const float4*>(bee + h);
  const float w0s[4] = {w0.x, w0.y, w0.z, w0.w};
  const float w1s[4] = {w1.x, w1.y, w1.z, w1.w};
  const float bs[4]  = {bb.x, bb.y, bb.z, bb.w};

  for (int wid = blockIdx.x * (blockDim.x >> 5) + (threadIdx.x >> 5);
       wid < E_DIM; wid += wpg) {
    const int s = src[wid];
    const int d = dst[wid];
    // prefetch next iteration's gather row (global_prefetch_b8)
    const int nxt = wid + wpg;
    if (nxt < E_DIM) {
      __builtin_prefetch(x + (size_t)src[nxt] * H_DIM + h, 0, 3);
    }
    const float ea0 = ea[2 * (size_t)wid];
    const float ea1 = ea[2 * (size_t)wid + 1];
    const float4 xv = *reinterpret_cast<const float4*>(x + (size_t)s * H_DIM + h);
    const float xs[4] = {xv.x, xv.y, xv.z, xv.w};

    float msg[4], lg[4];
    #pragma unroll
    for (int j = 0; j < 4; ++j) {
      float ev = ea0 * w0s[j] + ea1 * w1s[j] + bs[j];   // edge encoder inline
      msg[j] = fmaxf(xs[j] + ev, 0.0f) + EPS_MSG;
      lg[j]  = msg[j] * tv;
    }
    const size_t base = (size_t)d * H_DIM + h;
    if (pass == 0) {
      #pragma unroll
      for (int j = 0; j < 4; ++j) atomicMaxFloat(&m[base + j], lg[j]);
    } else if (pass == 1) {
      #pragma unroll
      for (int j = 0; j < 4; ++j)
        atomicAdd(&den[base + j], __expf(lg[j] - m[base + j]));
    } else {
      #pragma unroll
      for (int j = 0; j < 4; ++j) {
        float ex = __expf(lg[j] - m[base + j]);
        atomicAdd(&agg[base + j], msg[j] * ex / (den[base + j] + 1e-16f));
      }
    }
  }
}

__global__ void init_seg_kernel(float* __restrict__ m, float* __restrict__ den,
                                float* __restrict__ agg, int n) {
  int i = blockIdx.x * blockDim.x + threadIdx.x;
  if (i < n) { m[i] = -1e30f; den[i] = 0.0f; agg[i] = 0.0f; }
}

__global__ void fix_max_kernel(float* __restrict__ m, int n) {
  int i = blockIdx.x * blockDim.x + threadIdx.x;
  if (i < n && m[i] <= -1e29f) m[i] = 0.0f;   // empty-segment guard
}

__global__ void add_kernel(float* __restrict__ out, const float* __restrict__ a,
                           const float* __restrict__ b, int n) {
  int i = blockIdx.x * blockDim.x + threadIdx.x;
  if (i < n) out[i] = a[i] + b[i];
}

// ---------------------------------------------------------------------------
// LayerNorm (+optional ReLU), one block per row of width W, strided output
// ---------------------------------------------------------------------------
__global__ void ln_relu_kernel(const float* __restrict__ in, float* __restrict__ out,
                               const float* __restrict__ g, const float* __restrict__ b,
                               int W, int ostride, int ooffset, int relu)
{
  __shared__ float sh[32];
  const int n = blockIdx.x;
  const int c = threadIdx.x;
  float v  = in[(size_t)n * W + c];
  float mu = block_reduce_sum(v, sh) / (float)W;
  float dv = v - mu;
  float var = block_reduce_sum(dv * dv, sh) / (float)W;
  float o = dv * rsqrtf(var + 1e-5f) * g[c] + b[c];
  if (relu) o = fmaxf(o, 0.0f);
  out[(size_t)n * ostride + ooffset + c] = o;
}

// ---------------------------------------------------------------------------
// T=2 channel attention with gamma-gated residual; one 128-thread block/node
// ---------------------------------------------------------------------------
__global__ __launch_bounds__(128) void attn_kernel(
    const float* __restrict__ emb,
    const float* __restrict__ Wq, const float* __restrict__ Wk,
    const float* __restrict__ Wv, const float* __restrict__ gptr,
    float* __restrict__ out)
{
  __shared__ float sh[32];
  const int n = blockIdx.x;
  const int h = threadIdx.x;
  float x0 = emb[(size_t)n * 256 + h];
  float x1 = emb[(size_t)n * 256 + 128 + h];
  float q0 = Wq[0] * x0 + Wq[1] * x1, q1 = Wq[2] * x0 + Wq[3] * x1;
  float k0 = Wk[0] * x0 + Wk[1] * x1, k1 = Wk[2] * x0 + Wk[3] * x1;
  float v0 = Wv[0] * x0 + Wv[1] * x1, v1 = Wv[2] * x0 + Wv[3] * x1;
  float s00 = block_reduce_sum(q0 * k0, sh);
  float s01 = block_reduce_sum(q0 * k1, sh);
  float s10 = block_reduce_sum(q1 * k0, sh);
  float s11 = block_reduce_sum(q1 * k1, sh);
  float m0 = fmaxf(s00, s01), m1 = fmaxf(s10, s11);
  float e00 = __expf(s00 - m0), e01 = __expf(s01 - m0);
  float e10 = __expf(s10 - m1), e11 = __expf(s11 - m1);
  float r0 = e00 + e01, r1 = e10 + e11;
  float a00 = e00 / r0, a01 = e01 / r0, a10 = e10 / r1, a11 = e11 / r1;
  float gma = gptr[0];
  out[(size_t)n * 256 + h]       = gma * (a00 * v0 + a01 * v1) + x0;
  out[(size_t)n * 256 + 128 + h] = gma * (a10 * v0 + a11 * v1) + x1;
}

// ---------------------------------------------------------------------------
// node head: [N,256] @ [256,3] + log_softmax (one thread per node)
// ---------------------------------------------------------------------------
__global__ void node_head_kernel(const float* __restrict__ hidden,
                                 const float* __restrict__ W,
                                 const float* __restrict__ b,
                                 float* __restrict__ out)
{
  int n = blockIdx.x * blockDim.x + threadIdx.x;
  if (n >= N_NODES) return;
  float l0 = b[0], l1 = b[1], l2 = b[2];
  const float* hr = hidden + (size_t)n * 256;
  for (int c = 0; c < 256; ++c) {
    float hv = hr[c];
    l0 += hv * W[c * CN_DIM + 0];
    l1 += hv * W[c * CN_DIM + 1];
    l2 += hv * W[c * CN_DIM + 2];
  }
  float mx = fmaxf(l0, fmaxf(l1, l2));
  float ls = mx + __logf(__expf(l0 - mx) + __expf(l1 - mx) + __expf(l2 - mx));
  out[(size_t)n * CN_DIM + 0] = l0 - ls;
  out[(size_t)n * CN_DIM + 1] = l1 - ls;
  out[(size_t)n * CN_DIM + 2] = l2 - ls;
}

// ---------------------------------------------------------------------------
// graph pooling + head
// ---------------------------------------------------------------------------
__global__ void zero_kernel(float* __restrict__ p, int n) {
  int i = blockIdx.x * blockDim.x + threadIdx.x;
  if (i < n) p[i] = 0.0f;
}

__global__ __launch_bounds__(256) void pool_kernel(const float* __restrict__ g,
                                                   float* __restrict__ pooled)
{
  const int c    = threadIdx.x;
  const int base = blockIdx.x * 64;
  float acc = 0.0f;
  for (int r = 0; r < 64; ++r) {
    int n = base + r;
    if (n < N_NODES) acc += g[(size_t)n * 256 + c];
  }
  atomicAdd(&pooled[c], acc);
}

__global__ __launch_bounds__(256) void graph_head_kernel(
    const float* __restrict__ pooled,
    const float* __restrict__ W1, const float* __restrict__ b1,
    const float* __restrict__ W2, const float* __restrict__ b2,
    float* __restrict__ out)
{
  __shared__ float gp[256];
  __shared__ float h1[256];
  __shared__ float oo[CG_DIM];
  const int c = threadIdx.x;
  gp[c] = pooled[c] * (1.0f / (float)N_NODES);
  __syncthreads();
  float acc = b1[c];
  for (int j = 0; j < 256; ++j) acc += gp[j] * W1[j * 256 + c];
  h1[c] = fmaxf(acc, 0.0f);
  __syncthreads();
  if (c < CG_DIM) {
    float o = b2[c];
    for (int j = 0; j < 256; ++j) o += h1[j] * W2[j * CG_DIM + c];
    oo[c] = o;
  }
  __syncthreads();
  if (c == 0) {
    float mx = fmaxf(oo[0], oo[1]);
    float ls = mx + __logf(__expf(oo[0] - mx) + __expf(oo[1] - mx));
    out[0] = oo[0] - ls;
    out[1] = oo[1] - ls;
  }
}

// ---------------------------------------------------------------------------
// orchestration
// ---------------------------------------------------------------------------
extern "C" void kernel_launch(void* const* d_in, const int* in_sizes, int n_in,
                              void* d_out, int out_size, void* d_ws, size_t ws_size,
                              hipStream_t stream) {
  (void)in_sizes; (void)n_in; (void)out_size; (void)ws_size;
  const float* node_feature = (const float*)d_in[0];
  const int*   edge_index   = (const int*)  d_in[1];
  const float* edge_attr    = (const float*)d_in[2];
  const float* Wne  = (const float*)d_in[3];
  const float* bne  = (const float*)d_in[4];
  const float* Wee  = (const float*)d_in[5];
  const float* bee  = (const float*)d_in[6];
  const float* W1   = (const float*)d_in[7];
  const float* b1   = (const float*)d_in[8];
  const float* g1   = (const float*)d_in[9];
  const float* be1  = (const float*)d_in[10];
  const float* W2   = (const float*)d_in[11];
  const float* b2   = (const float*)d_in[12];
  const float* dg   = (const float*)d_in[13];
  const float* db   = (const float*)d_in[14];
  const float* tpar = (const float*)d_in[15];
  const float* Wq_n = (const float*)d_in[16];
  const float* Wk_n = (const float*)d_in[17];
  const float* Wv_n = (const float*)d_in[18];
  const float* gm_n = (const float*)d_in[19];
  const float* Wn1  = (const float*)d_in[20];
  const float* bn1  = (const float*)d_in[21];
  const float* Wn2  = (const float*)d_in[22];
  const float* bn2  = (const float*)d_in[23];
  const float* Wq_g = (const float*)d_in[24];
  const float* Wk_g = (const float*)d_in[25];
  const float* Wv_g = (const float*)d_in[26];
  const float* gm_g = (const float*)d_in[27];
  const float* Wg1  = (const float*)d_in[28];
  const float* bg1  = (const float*)d_in[29];
  const float* Wg2  = (const float*)d_in[30];
  const float* bg2  = (const float*)d_in[31];

  float* ws = (float*)d_ws;
  const size_t NH = (size_t)N_NODES * H_DIM;
  float* x0     = ws;
  float* x      = ws + 1 * NH;
  float* hbuf   = ws + 2 * NH;
  float* res    = ws + 3 * NH;
  float* mbuf   = ws + 4 * NH;
  float* den    = ws + 5 * NH;
  float* agg    = ws + 6 * NH;
  float* mlp1   = ws + 7 * NH;   // 2*NH
  float* emb    = ws + 9 * NH;   // 2*NH
  float* an     = ws + 11 * NH;  // 2*NH
  float* hidden = ws + 13 * NH;  // 2*NH
  float* ag     = ws + 15 * NH;  // 2*NH
  float* pooled = ws + 17 * NH;  // 256

  const int NHi       = N_NODES * H_DIM;
  const int ew_blocks = 2048;                     // persistent: 16384 waves
  const dim3 gemmH (2, (N_NODES + BM - 1) / BM);  // Nc = 128
  const dim3 gemm2H(4, (N_NODES + BM - 1) / BM);  // Nc = 256

  // ---- node encoder (shared by both transforms): x0 = nf @ Wne + bne
  gemm_wmma_kernel<<<gemmH, 256, 0, stream>>>(
      node_feature, Wne, bne, x0, N_NODES, H_DIM, F_DIM, 0, 0);

  for (int ti = 0; ti < T_DIM; ++ti) {
    const int*   srcp = edge_index + (size_t)ti * 2 * E_DIM;
    const int*   dstp = srcp + E_DIM;
    const float* eap  = edge_attr + (size_t)ti * E_DIM * 2;

    for (int l = 0; l < L_DIM; ++l) {
      const float* inx = (l == 0) ? x0 : hbuf;
      if (l > 0) {
        // res+ pre-activation: h = relu(LN(x, dg[l], db[l]))
        ln_relu_kernel<<<N_NODES, H_DIM, 0, stream>>>(
            x, hbuf, dg + l * H_DIM, db + l * H_DIM, H_DIM, H_DIM, 0, 1);
      }
      // segment softmax aggregation
      init_seg_kernel<<<(NHi + 255) / 256, 256, 0, stream>>>(mbuf, den, agg, NHi);
      edge_pass_kernel<<<ew_blocks, 256, 0, stream>>>(
          0, inx, srcp, dstp, eap, Wee, bee, tpar, l, mbuf, den, agg);
      fix_max_kernel<<<(NHi + 255) / 256, 256, 0, stream>>>(mbuf, NHi);
      edge_pass_kernel<<<ew_blocks, 256, 0, stream>>>(
          1, inx, srcp, dstp, eap, Wee, bee, tpar, l, mbuf, den, agg);
      edge_pass_kernel<<<ew_blocks, 256, 0, stream>>>(
          2, inx, srcp, dstp, eap, Wee, bee, tpar, l, mbuf, den, agg);
      // residual with root + MLP
      add_kernel<<<(NHi + 255) / 256, 256, 0, stream>>>(res, agg, inx, NHi);
      gemm_wmma_kernel<<<gemm2H, 256, 0, stream>>>(
          res, W1 + (size_t)l * H_DIM * 2 * H_DIM, b1 + l * 2 * H_DIM,
          mlp1, N_NODES, 2 * H_DIM, H_DIM, 0, 0);
      ln_relu_kernel<<<N_NODES, 2 * H_DIM, 0, stream>>>(
          mlp1, mlp1, g1 + l * 2 * H_DIM, be1 + l * 2 * H_DIM,
          2 * H_DIM, 2 * H_DIM, 0, 1);
      // layer 0: x = conv(x0); layers>=1: x += conv(h)
      gemm_wmma_kernel<<<gemmH, 256, 0, stream>>>(
          mlp1, W2 + (size_t)l * 2 * H_DIM * H_DIM, b2 + l * H_DIM,
          x, N_NODES, H_DIM, 2 * H_DIM, 0, (l == 0) ? 0 : 1);
    }
    // final per-transform: emb[:, ti, :] = relu(LN(x, dg[0], db[0]))
    ln_relu_kernel<<<N_NODES, H_DIM, 0, stream>>>(
        x, emb, dg, db, H_DIM, T_DIM * H_DIM, ti * H_DIM, 1);
  }

  // ---- node classification path
  attn_kernel<<<N_NODES, 128, 0, stream>>>(emb, Wq_n, Wk_n, Wv_n, gm_n, an);
  gemm_wmma_kernel<<<gemm2H, 256, 0, stream>>>(
      an, Wn1, bn1, hidden, N_NODES, 256, 256, 1, 0);
  node_head_kernel<<<(N_NODES + 255) / 256, 256, 0, stream>>>(
      hidden, Wn2, bn2, (float*)d_out);

  // ---- graph classification path
  attn_kernel<<<N_NODES, 128, 0, stream>>>(emb, Wq_g, Wk_g, Wv_g, gm_g, ag);
  zero_kernel<<<1, 256, 0, stream>>>(pooled, 256);
  pool_kernel<<<(N_NODES + 63) / 64, 256, 0, stream>>>(ag, pooled);
  graph_head_kernel<<<1, 256, 0, stream>>>(
      pooled, Wg1, bg1, Wg2, bg2, (float*)d_out + (size_t)N_NODES * CN_DIM);
}
